// MultiRNN_65661460021478
// MI455X (gfx1250) — matrix operand order repaired and checked by reading, not compiled, not run on hardware
//
#include <hip/hip_runtime.h>

typedef __bf16 bf16;
typedef __attribute__((ext_vector_type(16))) __bf16 v16bf;
typedef __attribute__((ext_vector_type(8)))  __bf16 v8bf;
typedef __attribute__((ext_vector_type(8)))  float  v8f;
typedef __attribute__((ext_vector_type(4)))  float  v4f;

#define NSAMP  131072
#define NFRM   128
#define CPDIM  16
#define NVOICE 8
#define NBLK   3
#define NCPI   512
#define NVERB  8
#define NEVENT 64
#define MAXACT 32
#define PADS   2048                 // front zero-pad of samples (16*NFRM)
#define RVLEN  (NSAMP + 64)         // reversed-verb phase copy length
#define SPLEN  (PADS + NSAMP + 64)  // padded samples length (bf16)
#define DCONST (NSAMP + 16)         // reversal constant, multiple of 8

__device__ __forceinline__ float sigmf(float x){ return 1.0f/(1.0f+__expf(-x)); }

__device__ __forceinline__ v8bf ld8(const bf16* p){ return *(const v8bf*)p; }

__device__ __forceinline__ v16bf cat16(v8bf a, v8bf b){
  v16bf r;
#pragma unroll
  for (int i=0;i<8;i++){ r[i]=a[i]; r[i+8]=b[i]; }
  return r;
}

__device__ __forceinline__ v8f wmma_bf(v16bf a, v16bf b, v8f c){
  // v_wmma_f32_16x16x32_bf16, f32 accumulate
  return __builtin_amdgcn_wmma_f32_16x16x32_bf16(false, a, false, b, (short)0, c, false, false);
}

// ---------------------------------------------------------------------------
// Static prep: pack audio maps (K = c*16+d, padded to 64) as bf16
__global__ void k_abf(const float* __restrict__ am, bf16* __restrict__ Abf){
  int idx = blockIdx.x*256 + threadIdx.x;
  if (idx >= NVOICE*1024*64) return;
  int cd = idx & 63, w = (idx>>6)&1023, v = idx>>16;
  float val = 0.0f;
  if (cd < 48){ int c = cd>>4, d = cd&15; val = am[(((size_t)(v*NBLK+c))*1024 + w)*CPDIM + d]; }
  Abf[idx] = (bf16)val;
}

// Static prep: 8 phase-shifted reversed copies of each verb, zero aprons.
// rvp[r][p][y] = verb_r[DCONST + p - y] (0 outside [0,NSAMP))
__global__ void k_rvp(const float* __restrict__ verbs, bf16* __restrict__ rvp){
  int idx = blockIdx.x*256 + threadIdx.x;
  if (idx >= NVERB*8*RVLEN) return;
  int y  = idx % RVLEN;
  int pr = idx / RVLEN;
  int p  = pr & 7, r = pr >> 3;
  int k  = DCONST + p - y;
  float val = (k >= 0 && k < NSAMP) ? verbs[(size_t)r*NSAMP + k] : 0.0f;
  rvp[idx] = (bf16)val;
}

// ---------------------------------------------------------------------------
// Per-event prep: argmaxes, softmax(room_mix), |amp|, cp normalize + top-32
__global__ void k_event_prep(const float* __restrict__ voice, const float* __restrict__ cpc,
                             const float* __restrict__ amps, const float* __restrict__ room,
                             const float* __restrict__ rmix, const float* __restrict__ cp_items,
                             int* __restrict__ pi, float* __restrict__ pf,
                             float* __restrict__ cp_sp){
  int e = blockIdx.x, tid = threadIdx.x; // 256 threads
  __shared__ float sv[2048];
  __shared__ float red[256];
  __shared__ int   redi[256];
  if (tid == 0){
    int vi=0; float bv=voice[e*8];
    for (int k=1;k<8;k++){ float x=voice[e*8+k]; if (x>bv){bv=x;vi=k;} }
    pi[e*4+0]=vi;
    int ri=0; float br=room[e*8];
    for (int k=1;k<8;k++){ float x=room[e*8+k]; if (x>br){br=x;ri=k;} }
    pi[e*4+1]=ri;
    pf[e*8+0]=fabsf(amps[e]);
    float r0=rmix[e*2+0], r1=rmix[e*2+1];
    float mx=fmaxf(r0,r1);
    float e0=__expf(r0-mx), e1=__expf(r1-mx), z=e0+e1;
    pf[e*8+1]=e0/z; pf[e*8+2]=e1/z;
  }
  // argmax over 512 cp choices (first-index tie-break)
  float bv = -3.4e38f; int bi = 0;
  for (int i=tid;i<NCPI;i+=256){ float x=cpc[e*NCPI+i]; if (x>bv){bv=x;bi=i;} }
  red[tid]=bv; redi[tid]=bi; __syncthreads();
  for (int s=128;s>0;s>>=1){
    if (tid<s){
      float ov=red[tid+s]; int oi=redi[tid+s];
      if (ov>red[tid] || (ov==red[tid] && oi<redi[tid])){ red[tid]=ov; redi[tid]=oi; }
    }
    __syncthreads();
  }
  int ci = redi[0];
  __syncthreads();
  float part = 0.0f;
  for (int i=tid;i<2048;i+=256){ float x=cp_items[(size_t)ci*2048+i]; sv[i]=x; part+=x; }
  red[tid]=part; __syncthreads();
  for (int s=128;s>0;s>>=1){ if (tid<s) red[tid]+=red[tid+s]; __syncthreads(); }
  float inv = 1.0f/(red[0]+1e-8f);
  __syncthreads();
  for (int i=tid;i<2048;i+=256) sv[i]*=inv;
  __syncthreads();
  // top-32 via rank counting (matches lax.top_k first-index tie-break)
  for (int i=tid;i<2048;i+=256){
    float v = sv[i]; int rank = 0;
    for (int j=0;j<2048;j++){ float u=sv[j]; rank += (u>v || (u==v && j<i)) ? 1 : 0; }
    cp_sp[(size_t)e*2048+i] = (rank < MAXACT) ? v : 0.0f;
  }
}

// ---------------------------------------------------------------------------
// Recurrent 3-block chain per event; emits Zt[e][f][cd] = mix_c * cp_out (bf16)
__global__ void k_chain(const float* __restrict__ w1, const float* __restrict__ w2,
                        const float* __restrict__ decays, const float* __restrict__ gains,
                        const float* __restrict__ mixp, const float* __restrict__ cp_sp,
                        const int* __restrict__ pi, bf16* __restrict__ Zt){
  int e = blockIdx.x;
  int tid = threadIdx.x;  // 128 threads, f = tid
  int v = pi[e*4+0];
  __shared__ float wc[CPDIM][NFRM];
  __shared__ float xx[CPDIM][NFRM];
  __shared__ float yy[CPDIM][NFRM];
  __shared__ float sW1[CPDIM*CPDIM], sW2[CPDIM*CPDIM];
  __shared__ float sdec[CPDIM], sg[CPDIM], smix[NBLK];
  for (int i=tid;i<CPDIM*NFRM;i+=128) wc[i/NFRM][i%NFRM] = cp_sp[(size_t)e*2048+i];
  if (tid == 0){
    float a=mixp[v*NBLK+0], b=mixp[v*NBLK+1], c=mixp[v*NBLK+2];
    float mx=fmaxf(a,fmaxf(b,c));
    float ea=__expf(a-mx), eb=__expf(b-mx), ec=__expf(c-mx), z=ea+eb+ec;
    smix[0]=ea/z; smix[1]=eb/z; smix[2]=ec/z;
  }
  __syncthreads();
  int f = tid;
  for (int blk=0; blk<NBLK; ++blk){
    const float* W1 = w1 + (size_t)(v*NBLK+blk)*CPDIM*CPDIM;
    const float* W2 = w2 + (size_t)(v*NBLK+blk)*CPDIM*CPDIM;
    for (int i=tid;i<CPDIM*CPDIM;i+=128){ sW1[i]=W1[i]; sW2[i]=W2[i]; }
    if (tid < CPDIM){
      sdec[tid] = 1e-12f + 0.5f + sigmf(decays[(v*NBLK+blk)*CPDIM+tid])*0.5f;
      sg[tid]   = sigmf(gains[(v*NBLK+blk)*CPDIM+tid])*5.0f;
    }
    __syncthreads();
    float inp[CPDIM];
#pragma unroll
    for (int d=0; d<CPDIM; ++d) inp[d] = fmaxf(wc[d][f], 0.0f);
#pragma unroll
    for (int d=0; d<CPDIM; ++d){
      float s = 0.0f;
#pragma unroll
      for (int k=0;k<CPDIM;k++) s += sW1[d*CPDIM+k]*inp[k];
      xx[d][f] = s;
    }
    __syncthreads();
    // env FFT-conv == exact 1-pole IIR: y[f] = dec*(y[f-1] + x[f])
    if (tid < CPDIM){
      float dec = sdec[tid], y = 0.0f;
      for (int ff=0; ff<NFRM; ++ff){ y = dec*(y + xx[tid][ff]); yy[tid][ff] = y; }
    }
    __syncthreads();
#pragma unroll
    for (int d=0; d<CPDIM; ++d){
      float s = xx[d][f];
#pragma unroll
      for (int k=0;k<CPDIM;k++) s += sW2[d*CPDIM+k]*yy[k][f];
      float o = tanhf(s*sg[d]);
      wc[d][f] = o;
      Zt[((size_t)e*NFRM+f)*64 + blk*CPDIM + d] = (bf16)(smix[blk]*o);
    }
    __syncthreads();
  }
  for (int d=0; d<16; ++d) Zt[((size_t)e*NFRM+f)*64 + 48 + d] = (bf16)0.0f; // K pad
}

// ---------------------------------------------------------------------------
// mixed[e, f*1024+w] = sum_cd Abf[v][w][cd] * Zt[e][f][cd]  (K=64 bf16 WMMA)
__global__ void k_audio(const bf16* __restrict__ Abf, const bf16* __restrict__ Zt,
                        const int* __restrict__ pi, float* __restrict__ mixed){
  int wid = blockIdx.x*4 + (threadIdx.x>>5);
  int l = threadIdx.x & 31;
  int e = wid >> 9;
  int t = wid & 511;
  int W0 = (t>>3)<<4;
  int F0 = (t&7)<<4;
  int v = pi[e*4+0];
  int n = l & 15, hi = l >> 4;
  v8f acc = {};
#pragma unroll
  for (int q=0; q<2; ++q){
    // A[m,kk]: runs at kk offsets {8hi..8hi+7} and {16+8hi..}
    const bf16* ap = Abf + ((size_t)(v*1024 + W0 + n))*64 + 32*q + 8*hi;
    // B[kk,nn]: runs at kk offsets {8hi..8hi+7} and {16+8hi..}
    const bf16* bp = Zt  + ((size_t)(e*NFRM + F0 + n))*64 + 32*q + 8*hi;
    v16bf A = cat16(ld8(ap), ld8(ap+16));
    v16bf B = cat16(ld8(bp), ld8(bp+16));
    acc = wmma_bf(A, B, acc);
  }
  float* out = mixed + (size_t)e*NSAMP + (size_t)(F0+n)*1024 + W0 + 8*hi;
  v4f lo = {acc[0],acc[1],acc[2],acc[3]};
  v4f hh = {acc[4],acc[5],acc[6],acc[7]};
  *(v4f*)(out)   = lo;
  *(v4f*)(out+4) = hh;
}

// ---------------------------------------------------------------------------
__global__ void k_maxabs(const float* __restrict__ mixed, float* __restrict__ pf){
  int e = blockIdx.x, tid = threadIdx.x;
  __shared__ float red[256];
  float m = 0.0f;
  const float* me = mixed + (size_t)e*NSAMP;
  for (int i=tid;i<NSAMP;i+=256) m = fmaxf(m, fabsf(me[i]));
  red[tid]=m; __syncthreads();
  for (int s=128;s>0;s>>=1){ if (tid<s) red[tid]=fmaxf(red[tid],red[tid+s]); __syncthreads(); }
  if (tid==0) pf[e*8+3] = 1.0f/(red[0]+1e-8f);
}

// normalized samples -> bf16, zero front/back pads
__global__ void k_sp(const float* __restrict__ mixed, const float* __restrict__ pf,
                     bf16* __restrict__ sp){
  long long idx = (long long)blockIdx.x*256 + threadIdx.x;
  if (idx >= (long long)NEVENT*SPLEN) return;
  int e = (int)(idx / SPLEN);
  int x = (int)(idx % SPLEN);
  float val = 0.0f;
  if (x >= PADS && x < PADS+NSAMP) val = mixed[(size_t)e*NSAMP + (x-PADS)] * pf[e*8+3];
  sp[idx] = (bf16)val;
}

// ---------------------------------------------------------------------------
// Reverb conv as Toeplitz-tiled bf16 WMMA GEMM; fuses wet/dry mix + amp.
// Tile: t = T0 + 16*(m+16g) + n ; A = sp (aligned), B = phase-copied reversed verb.
__global__ void k_conv(const bf16* __restrict__ sp, const bf16* __restrict__ rvp,
                       const int* __restrict__ pi, const float* __restrict__ pf,
                       float* __restrict__ mixed){
  int wid = blockIdx.x*4 + (threadIdx.x>>5);
  int l = threadIdx.x & 31;
  int e  = wid >> 6;
  int wt = wid & 63;
  int T0 = wt << 11;            // 2048 samples per wave
  int n = l & 15, hi = l >> 4;
  int room = pi[e*4+1];
  float amp = pf[e*8+0], vm0 = pf[e*8+1], vm1 = pf[e*8+2], scale = pf[e*8+3];
  const bf16* spe = sp + (size_t)e*SPLEN;
  int p = n & 7, h = n >> 3;
  const bf16* rv = rvp + ((size_t)(room*8 + p))*RVLEN;
  int ybase = DCONST - (T0 + PADS) + 8*hi - 8*h;   // aligned mod 8 by construction
  v8f zero = {};
  v8f acc[8];
#pragma unroll
  for (int g=0; g<8; ++g) acc[g] = zero;
  int nsteps = ((T0 + PADS)>>5) + 1;
  for (int s=0; s<nsteps; ++s){
    int S0 = s<<5;
    const bf16* bp = rv + ybase + S0;
    if ((s & 15) == 0){
      __builtin_prefetch(bp + 512, 0, 1);              // global_prefetch_b8
      __builtin_prefetch(spe + S0 + 16*n + 2048, 0, 1);
    }
    v16bf B = cat16(ld8(bp), ld8(bp+16));               // shared across 8 tiles
    const bf16* abase = spe + S0 + 16*n + 8*hi;
#pragma unroll
    for (int g=0; g<8; ++g){
      const bf16* ap = abase + 256*g;
      v16bf A = cat16(ld8(ap), ld8(ap+16));
      acc[g] = wmma_bf(A, B, acc[g]);
    }
  }
  float* me = mixed + (size_t)e*NSAMP;
#pragma unroll
  for (int g=0; g<8; ++g){
#pragma unroll
    for (int r=0; r<8; ++r){
      int t = T0 + 16*((r + 8*hi) + 16*g) + n;
      float dry = me[t]*scale;
      me[t] = amp*(vm0*acc[g][r] + vm1*dry);  // in-place: each t touched once
    }
  }
}

// ---------------------------------------------------------------------------
// Dirac comb: out[t] = sum_{k<=t>>10} times[k] * final[t - 1024k]
__global__ void k_dirac(const float* __restrict__ fin, const float* __restrict__ times,
                        float* __restrict__ out){
  int gid = blockIdx.x*256 + threadIdx.x;
  int e = gid >> 17;
  int t = gid & (NSAMP-1);
  int f = t >> 10;              // uniform per block -> scalar times[k]
  const float* fb = fin + (size_t)e*NSAMP;
  const float* tm = times + e*NFRM;
  float acc = 0.0f;
  for (int k=0; k<=f; ++k) acc += tm[k]*fb[t - (k<<10)];
  out[gid] = acc;
}

// ---------------------------------------------------------------------------
extern "C" void kernel_launch(void* const* d_in, const int* in_sizes, int n_in,
                              void* d_out, int out_size, void* d_ws, size_t ws_size,
                              hipStream_t stream) {
  (void)in_sizes; (void)n_in; (void)out_size; (void)ws_size;
  const float* voice    = (const float*)d_in[0];
  const float* cpc      = (const float*)d_in[1];
  const float* amps     = (const float*)d_in[2];
  const float* room     = (const float*)d_in[3];
  const float* rmix     = (const float*)d_in[4];
  const float* times    = (const float*)d_in[5];
  const float* cp_items = (const float*)d_in[6];
  const float* verbs    = (const float*)d_in[7];
  const float* w1       = (const float*)d_in[8];
  const float* w2       = (const float*)d_in[9];
  const float* am       = (const float*)d_in[10];
  const float* decays   = (const float*)d_in[11];
  const float* gains    = (const float*)d_in[12];
  const float* mixp     = (const float*)d_in[13];

  char* p = (char*)d_ws;
  auto alloc = [&](size_t bytes)->void*{
    void* r = (void*)p; p += (bytes + 255) & ~(size_t)255; return r;
  };
  int*   pi    = (int*)  alloc((size_t)NEVENT*4*sizeof(int));
  float* pf    = (float*)alloc((size_t)NEVENT*8*sizeof(float));
  float* cp_sp = (float*)alloc((size_t)NEVENT*2048*sizeof(float));
  bf16*  Zt    = (bf16*) alloc((size_t)NEVENT*NFRM*64*sizeof(bf16));
  bf16*  Abf   = (bf16*) alloc((size_t)NVOICE*1024*64*sizeof(bf16));
  bf16*  rvp   = (bf16*) alloc((size_t)NVERB*8*RVLEN*sizeof(bf16));
  bf16*  spb   = (bf16*) alloc((size_t)NEVENT*SPLEN*sizeof(bf16));
  float* mixed = (float*)alloc((size_t)NEVENT*NSAMP*sizeof(float)); // also "final"

  k_abf   <<<(NVOICE*1024*64 + 255)/256, 256, 0, stream>>>(am, Abf);
  k_rvp   <<<(NVERB*8*RVLEN + 255)/256, 256, 0, stream>>>(verbs, rvp);
  k_event_prep<<<NEVENT, 256, 0, stream>>>(voice, cpc, amps, room, rmix, cp_items,
                                           pi, pf, cp_sp);
  k_chain <<<NEVENT, 128, 0, stream>>>(w1, w2, decays, gains, mixp, cp_sp, pi, Zt);
  k_audio <<<(NEVENT*512)/4, 128, 0, stream>>>(Abf, Zt, pi, mixed);
  k_maxabs<<<NEVENT, 256, 0, stream>>>(mixed, pf);
  {
    long long tot = (long long)NEVENT*SPLEN;
    k_sp <<<(int)((tot + 255)/256), 256, 0, stream>>>(mixed, pf, spb);
  }
  k_conv  <<<(NEVENT*64)/4, 128, 0, stream>>>(spb, rvp, pi, pf, mixed);
  k_dirac <<<(NEVENT*NSAMP)/256, 256, 0, stream>>>(mixed, times, (float*)d_out);
}
